// MCLSTMReset_70660801954474
// MI455X (gfx1250) — matrix operand order repaired and checked by reading, not compiled, not run on hardware
//
#include <hip/hip_runtime.h>

// ---------------------------------------------------------------------------
// MCLSTMReset for MI455X (gfx1250): 2-layer LSTM scan, T=64, B=1024, H=IN=512.
// GEMMs run on bf16 WMMA (v_wmma_f32_16x16x32_bf16) with f32 accumulation.
// Weights converted once to bf16 (8 MB, L2-resident). Gates buffer (8 MB)
// round-trips through L2. Elementwise cell math stays f32.
// GEMM waves are register-blocked 2x2 (32x32 output per wave) so each A/B
// fragment load feeds two WMMAs: 8 b128 loads -> 4 WMMAs per K=32 step.
// ---------------------------------------------------------------------------

typedef __attribute__((ext_vector_type(16))) __bf16 v16bf;
typedef __attribute__((ext_vector_type(8)))  __bf16 v8bf;
typedef __attribute__((ext_vector_type(8)))  float  v8f;

#define TT 64
#define BB 1024
#define HH 512
#define G4 2048            // 4*H
#define BH (BB * HH)       // 524288

__device__ __forceinline__ float sigf(float x) {
    return 1.0f / (1.0f + __expf(-x));
}

// --------------------------------------------------------------------------
// One-time prep: convert 4 weight matrices [2048,512] f32 -> bf16, sum biases.
// --------------------------------------------------------------------------
__global__ __launch_bounds__(256)
void prep_kernel(const float* __restrict__ wih0, const float* __restrict__ whh0,
                 const float* __restrict__ wih1, const float* __restrict__ whh1,
                 const float* __restrict__ bih0, const float* __restrict__ bhh0,
                 const float* __restrict__ bih1, const float* __restrict__ bhh1,
                 __bf16* __restrict__ wbf, float* __restrict__ bias0,
                 float* __restrict__ bias1)
{
    const long NW = 4L * G4 * HH;                 // 4,194,304
    long idx = (long)blockIdx.x * 256 + threadIdx.x;
    if (idx < NW) {
        long which = idx / (G4 * HH);
        long off   = idx % (G4 * HH);
        const float* src = (which == 0) ? wih0 : (which == 1) ? whh0
                         : (which == 2) ? wih1 : whh1;
        wbf[idx] = (__bf16)src[off];
    } else {
        long r = idx - NW;
        if (r < G4)           bias0[r]        = bih0[r]        + bhh0[r];
        else if (r < 2 * G4)  bias1[r - G4]   = bih1[r - G4]   + bhh1[r - G4];
    }
}

// --------------------------------------------------------------------------
// Per-step: apply reset/clear mask + noise injection to (h,c) of both layers,
// emit bf16 copies of masked h for the WMMA GEMMs, convert latent[t] to bf16.
// jax bool arrays are 1 byte/element -> unsigned char.
// --------------------------------------------------------------------------
__global__ __launch_bounds__(256)
void maskstep_kernel(const float* __restrict__ latent_t,      // [B,IN]
                     const float* __restrict__ noise_t,       // [L,B,H]
                     const unsigned char* __restrict__ reset_mask, // [B,T]
                     const unsigned char* __restrict__ clear_mask, // [B,T]
                     const float* __restrict__ std_bt,        // [B,T]
                     int t,
                     float* __restrict__ h_state,             // [2,B,H] f32
                     float* __restrict__ c_state,             // [2,B,H] f32
                     __bf16* __restrict__ xbf,                // [B,IN]
                     __bf16* __restrict__ h0m,                // [B,H]
                     __bf16* __restrict__ h1m)                // [B,H]
{
    int idx = blockIdx.x * 256 + threadIdx.x;   // [0, B*H)
    int b   = idx >> 9;

    xbf[idx] = (__bf16)latent_t[idx];

    bool  m = (reset_mask[b * TT + t] | clear_mask[b * TT + t]) != 0;
    float s = std_bt[b * TT + t];

    // layer 0
    {
        float sn = noise_t[idx] * s;
        float h  = (m ? 0.0f : h_state[idx]) + sn;
        float c  = (m ? 0.0f : c_state[idx]) + sn;
        c_state[idx] = c;
        h0m[idx]     = (__bf16)h;
    }
    // layer 1
    {
        int   j  = BH + idx;
        float sn = noise_t[j] * s;
        float h  = (m ? 0.0f : h_state[j]) + sn;
        float c  = (m ? 0.0f : c_state[j]) + sn;
        c_state[j] = c;
        h1m[idx]   = (__bf16)h;
    }
}

// --------------------------------------------------------------------------
// A fragment loader: 16x32 bf16 tile of A starting at row `m0`, K offset k0.
// Per ISA layout: lanes 0-15 (M=lane) VGPR0..3 = K hi*8..hi*8+7,
//                 VGPR4..7 = K 16+hi*8..16+hi*8+7 (two contiguous 16B runs).
// --------------------------------------------------------------------------
__device__ __forceinline__ v16bf load_afrag(const __bf16* __restrict__ arow,
                                            int k0, int hi)
{
    v8bf lo = *(const v8bf*)(arow + k0 + hi * 8);
    v8bf hi8 = *(const v8bf*)(arow + k0 + 16 + hi * 8);
    v16bf v;
#pragma unroll
    for (int i = 0; i < 8; ++i) { v[i] = lo[i]; v[i + 8] = hi8[i]; }
    return v;
}

// B fragment: col N = lane%16 (weight row), K order matches A (hi*8.., 16+hi*8..).
__device__ __forceinline__ v16bf load_bfrag(const __bf16* __restrict__ wrow,
                                            int k0, int hi)
{
    v8bf lo = *(const v8bf*)(wrow + k0 + hi * 8);
    v8bf hi8 = *(const v8bf*)(wrow + k0 + 16 + hi * 8);
    v16bf v;
#pragma unroll
    for (int i = 0; i < 8; ++i) { v[i] = lo[i]; v[i + 8] = hi8[i]; }
    return v;
}

// --------------------------------------------------------------------------
// Fused dual GEMM: gates[B,4H] = A0[B,512] * W0[4H,512]^T + A1 * W1^T.
// Weights row-major [N][K] are exactly the WMMA B-fragment layout
// (lane = N, contiguous K per lane) -> no transpose loads needed.
// Block = 256 threads = 8 waves (2M x 4N), block tile 64(M) x 128(N).
// Each wave: 2x2 of 16x16 tiles, 4 accumulators; 4 WMMAs per 8 b128 loads.
// --------------------------------------------------------------------------
__global__ __launch_bounds__(256)
void gemm2_kernel(const __bf16* __restrict__ A0, const __bf16* __restrict__ W0,
                  const __bf16* __restrict__ A1, const __bf16* __restrict__ W1,
                  float* __restrict__ gates)
{
    const int lane = threadIdx.x & 31;
    const int wave = threadIdx.x >> 5;
    const int bm = blockIdx.y * 64 + (wave >> 2) * 32;   // wave M base (2 tiles)
    const int bn = blockIdx.x * 128 + (wave & 3) * 32;   // wave N base (2 tiles)
    const int lm = lane & 15;
    const int hi = lane >> 4;

    v8f acc00 = {}, acc01 = {}, acc10 = {}, acc11 = {};

#pragma unroll
    for (int p = 0; p < 2; ++p) {
        const __bf16* __restrict__ A = p ? A1 : A0;
        const __bf16* __restrict__ W = p ? W1 : W0;
        const __bf16* arow0 = A + (long)(bm + lm) * HH;       // M tile 0
        const __bf16* arow1 = A + (long)(bm + 16 + lm) * HH;  // M tile 1
        const __bf16* wrow0 = W + (long)(bn + lm) * HH;       // N tile 0
        const __bf16* wrow1 = W + (long)(bn + 16 + lm) * HH;  // N tile 1

#pragma unroll 2
        for (int k0 = 0; k0 < HH; k0 += 32) {
            // speculative prefetch of next K block (weights stream, L2-hot)
            __builtin_prefetch(wrow0 + k0 + 32, 0, 1);
            __builtin_prefetch(wrow1 + k0 + 32, 0, 1);

            v16bf a0 = load_afrag(arow0, k0, hi);
            v16bf a1 = load_afrag(arow1, k0, hi);
            v16bf b0 = load_bfrag(wrow0, k0, hi);
            v16bf b1 = load_bfrag(wrow1, k0, hi);

            acc00 = __builtin_amdgcn_wmma_f32_16x16x32_bf16(
                false, a0, false, b0, (short)0, acc00, false, false);
            acc01 = __builtin_amdgcn_wmma_f32_16x16x32_bf16(
                false, a0, false, b1, (short)0, acc01, false, false);
            acc10 = __builtin_amdgcn_wmma_f32_16x16x32_bf16(
                false, a1, false, b0, (short)0, acc10, false, false);
            acc11 = __builtin_amdgcn_wmma_f32_16x16x32_bf16(
                false, a1, false, b1, (short)0, acc11, false, false);
        }
    }

    // C/D layout: VGPR j -> M = tile_m + hi*8 + j, N = tile_n + lm
    float* o00 = gates + (long)(bm + hi * 8) * G4 + (bn + lm);
    float* o01 = o00 + 16;
    float* o10 = gates + (long)(bm + 16 + hi * 8) * G4 + (bn + lm);
    float* o11 = o10 + 16;
#pragma unroll
    for (int j = 0; j < 8; ++j) {
        o00[(long)j * G4] = acc00[j];
        o01[(long)j * G4] = acc01[j];
        o10[(long)j * G4] = acc10[j];
        o11[(long)j * G4] = acc11[j];
    }
}

// --------------------------------------------------------------------------
// LSTM cell pointwise: gate order i, f, g, o. Updates f32 state, emits bf16
// h for the next GEMM, optionally writes hidden output slice.
// --------------------------------------------------------------------------
__global__ __launch_bounds__(256)
void cell_kernel(const float* __restrict__ gates,   // [B,4H]
                 const float* __restrict__ bias,    // [4H]
                 float* __restrict__ c_state_l,     // [B,H]
                 float* __restrict__ h_state_l,     // [B,H]
                 __bf16* __restrict__ hbf,          // [B,H]
                 float* __restrict__ hidden_out)    // [B,H] or nullptr
{
    int idx = blockIdx.x * 256 + threadIdx.x;   // [0, B*H)
    int b   = idx >> 9;
    int k   = idx & 511;
    const float* g = gates + (long)b * G4;

    float gi = g[k]            + bias[k];
    float gf = g[HH + k]       + bias[HH + k];
    float gg = g[2 * HH + k]   + bias[2 * HH + k];
    float go = g[3 * HH + k]   + bias[3 * HH + k];

    float c = sigf(gf) * c_state_l[idx] + sigf(gi) * tanhf(gg);
    float h = sigf(go) * tanhf(c);

    c_state_l[idx] = c;
    h_state_l[idx] = h;
    hbf[idx]       = (__bf16)h;
    if (hidden_out) hidden_out[idx] = h;
}

// --------------------------------------------------------------------------
// Host-side orchestration (graph-capture safe: only kernel launches and
// async d2d copies on `stream`).
// --------------------------------------------------------------------------
extern "C" void kernel_launch(void* const* d_in, const int* in_sizes, int n_in,
                              void* d_out, int out_size, void* d_ws, size_t ws_size,
                              hipStream_t stream)
{
    (void)in_sizes; (void)n_in; (void)out_size; (void)ws_size;

    const float* latent   = (const float*)d_in[0];          // [T,B,IN]
    const float* h_t      = (const float*)d_in[1];          // [L,B,H]
    const float* c_t      = (const float*)d_in[2];          // [L,B,H]
    const unsigned char* reset_mask = (const unsigned char*)d_in[3]; // [1,B,T,1] bool
    const unsigned char* clear_mask = (const unsigned char*)d_in[4]; // [B,T] bool
    const float* std_bt   = (const float*)d_in[5];          // [B,T]
    const float* noise    = (const float*)d_in[6];          // [T,L,B,H]
    const float* w_ih0    = (const float*)d_in[7];
    const float* w_hh0    = (const float*)d_in[8];
    const float* b_ih0    = (const float*)d_in[9];
    const float* b_hh0    = (const float*)d_in[10];
    const float* w_ih1    = (const float*)d_in[11];
    const float* w_hh1    = (const float*)d_in[12];
    const float* b_ih1    = (const float*)d_in[13];
    const float* b_hh1    = (const float*)d_in[14];

    float* out = (float*)d_out;   // [T*B*H hidden][2*B*H h_f][2*B*H c_f]

    // ---- workspace carve-up (bytes) ----
    char* ws = (char*)d_ws;
    __bf16* wbf   = (__bf16*)(ws);                    // 4 x [2048,512] bf16 = 8 MiB
    __bf16* Wih0b = wbf;
    __bf16* Whh0b = wbf + 1 * G4 * HH;
    __bf16* Wih1b = wbf + 2 * G4 * HH;
    __bf16* Whh1b = wbf + 3 * G4 * HH;
    size_t off = 4L * G4 * HH * sizeof(__bf16);       // 8,388,608

    float* bias0 = (float*)(ws + off); off += G4 * sizeof(float);
    float* bias1 = (float*)(ws + off); off += G4 * sizeof(float);
    float* h_state = (float*)(ws + off); off += 2L * BH * sizeof(float);
    float* c_state = (float*)(ws + off); off += 2L * BH * sizeof(float);
    __bf16* xbf    = (__bf16*)(ws + off); off += (size_t)BH * sizeof(__bf16);
    __bf16* h0m    = (__bf16*)(ws + off); off += (size_t)BH * sizeof(__bf16);
    __bf16* h1m    = (__bf16*)(ws + off); off += (size_t)BH * sizeof(__bf16);
    __bf16* h0new  = (__bf16*)(ws + off); off += (size_t)BH * sizeof(__bf16);
    float* gates   = (float*)(ws + off); off += (size_t)BB * G4 * sizeof(float);

    // ---- one-time prep ----
    {
        long n = 4L * G4 * HH + 2L * G4;
        int blocks = (int)((n + 255) / 256);
        prep_kernel<<<blocks, 256, 0, stream>>>(w_ih0, w_hh0, w_ih1, w_hh1,
                                                b_ih0, b_hh0, b_ih1, b_hh1,
                                                wbf, bias0, bias1);
    }
    hipMemcpyAsync(h_state, h_t, 2L * BH * sizeof(float),
                   hipMemcpyDeviceToDevice, stream);
    hipMemcpyAsync(c_state, c_t, 2L * BH * sizeof(float),
                   hipMemcpyDeviceToDevice, stream);

    const dim3 gemm_grid(G4 / 128, BB / 64);  // 16 x 16 blocks, 8 waves each
    const int  ew_blocks = BH / 256;          // 2048 blocks

    // ---- sequential scan over T ----
    for (int t = 0; t < TT; ++t) {
        const float* latent_t = latent + (long)t * BB * HH;
        const float* noise_t  = noise  + (long)t * 2 * BH;

        maskstep_kernel<<<ew_blocks, 256, 0, stream>>>(
            latent_t, noise_t, reset_mask, clear_mask, std_bt, t,
            h_state, c_state, xbf, h0m, h1m);

        // layer 0: gates = x*Wih0^T + h0*Whh0^T
        gemm2_kernel<<<gemm_grid, 256, 0, stream>>>(xbf, Wih0b, h0m, Whh0b, gates);
        cell_kernel<<<ew_blocks, 256, 0, stream>>>(
            gates, bias0, c_state, h_state, h0new, nullptr);

        // layer 1: gates = h0new*Wih1^T + h1*Whh1^T
        gemm2_kernel<<<gemm_grid, 256, 0, stream>>>(h0new, Wih1b, h1m, Whh1b, gates);
        cell_kernel<<<ew_blocks, 256, 0, stream>>>(
            gates, bias1, c_state + BH, h_state + BH, h1m,
            out + (long)t * BH);
    }

    // ---- append final h/c state to output ----
    hipMemcpyAsync(out + (long)TT * BH, h_state,
                   2L * BH * sizeof(float), hipMemcpyDeviceToDevice, stream);
    hipMemcpyAsync(out + (long)TT * BH + 2L * BH, c_state,
                   2L * BH * sizeof(float), hipMemcpyDeviceToDevice, stream);
}
